// InterProceduralAbstractGraph_61297773248661
// MI455X (gfx1250) — matrix operand order repaired and checked by reading, not compiled, not run on hardware
//
#include <hip/hip_runtime.h>
#include <stdint.h>

// ---------------------------------------------------------------------------
// CDNA5 (gfx1250) implementation: all heavy GEMMs run on
// v_wmma_f32_16x16x32_bf16 (bf16 operands, f32 accumulate).
// GEMM is register-blocked: 32(M) x 64(N) per wave -> 8 WMMAs per K-step,
// each A-frag reused 4x and each B-frag 2x from registers.
// __launch_bounds__(64,1) + unroll 2 keeps the whole working set in VGPRs
// (previous round spilled to scratch inside the inner loop).
// ---------------------------------------------------------------------------

typedef __attribute__((ext_vector_type(16))) __bf16 v16bf;
typedef __attribute__((ext_vector_type(8)))  __bf16 v8bf;
typedef __attribute__((ext_vector_type(8)))  float  v8f;

#define GF_BIAS  1
#define GF_RELU  2
#define GF_ACC   4
#define GF_OUTBF 8

#define HGDIM 256   // GRU hidden per direction
#define G3    768   // 3*HG

static __device__ __forceinline__ __bf16 f2bf(float f) {
  unsigned u = __builtin_bit_cast(unsigned, f);
  u += 0x7FFFu + ((u >> 16) & 1u);          // round-to-nearest-even
  unsigned short h = (unsigned short)(u >> 16);
  return __builtin_bit_cast(__bf16, h);
}

// ------------------------- elementwise helpers ----------------------------

extern "C" __global__ void k_f32_to_bf16(const float* __restrict__ in,
                                         __bf16* __restrict__ out, long n) {
  long i = (long)blockIdx.x * blockDim.x + threadIdx.x;
  if (i < n) out[i] = f2bf(in[i]);
}

extern "C" __global__ void k_copy_f32(const float* __restrict__ in,
                                      float* __restrict__ out, long n) {
  long i = (long)blockIdx.x * blockDim.x + threadIdx.x;
  if (i < n) out[i] = in[i];
}

// ------------------------- degrees / successor ----------------------------

extern "C" __global__ void k_degrees(const int* __restrict__ row,
                                     const int* __restrict__ col,
                                     int* __restrict__ in_deg,
                                     int* __restrict__ out_deg,
                                     int* __restrict__ succ, int E) {
  int e = blockIdx.x * blockDim.x + threadIdx.x;
  if (e >= E) return;
  int r = row[e], c = col[e];
  atomicAdd(&out_deg[r], 1);
  atomicAdd(&in_deg[c], 1);
  succ[r] = c;   // valid where out_deg==1 (matches .at[row].set(col))
}

// merge input: abuf[n, 0:256] = bf16(x[n]), abuf[n, 256:512] = bf16(x[succ[n]])
extern "C" __global__ void k_build_merge_input(const float* __restrict__ x,
                                               const int* __restrict__ succ,
                                               __bf16* __restrict__ abuf, long n256) {
  long i = (long)blockIdx.x * blockDim.x + threadIdx.x;
  if (i >= n256) return;
  long n = i >> 8;
  int  d = (int)(i & 255);
  abuf[n * 512 + d]       = f2bf(x[n * 256 + d]);
  abuf[n * 512 + 256 + d] = f2bf(x[(long)succ[n] * 256 + d]);
}

// ------------------------- WMMA GEMM:  C = A * B^T ------------------------
// A: [M,K] bf16 row-major (lda); B: [N,K] bf16 row-major (ldb) (weights [out,in])
// Each wave computes a 32x64 block: 2 M-tiles x 4 N-tiles, 8 f32 accumulators.
// Epilogue: +bias, relu, accumulate-into-Cf, or emit bf16 to Cb.
// Requires M % 32 == 0 and N % 128 == 0 with blockDim=(32,2) (true here).

static __device__ __forceinline__ v16bf ld_a_frag(const __bf16* p) {
  const v8bf* ap = reinterpret_cast<const v8bf*>(p);
  v8bf a0 = ap[0];
  v8bf a1 = ap[2];                 // +16 elements
  v16bf r;
#pragma unroll
  for (int i = 0; i < 8; ++i) { r[i] = a0[i]; r[8 + i] = a1[i]; }
  return r;
}

static __device__ __forceinline__ v16bf ld_b_frag(const __bf16* p) {
  const v8bf* bp = reinterpret_cast<const v8bf*>(p);
  v8bf b0 = bp[0];
  v8bf b1 = bp[2];                 // +16 elements
  v16bf r;
#pragma unroll
  for (int i = 0; i < 8; ++i) { r[i] = b0[i]; r[8 + i] = b1[i]; }
  return r;
}

extern "C" __global__ void __launch_bounds__(64, 1)
k_gemm_wmma(const __bf16* __restrict__ A,
            const __bf16* __restrict__ B,
            const float* __restrict__ bias,
            float* __restrict__ Cf,
            __bf16* __restrict__ Cb,
            int M, int Nc, int K,
            int lda, int ldb, int ldc, int flags) {
  const int lane = threadIdx.x;                       // 0..31 (wave32)
  const int tn   = blockIdx.x * blockDim.y + threadIdx.y;  // 64-col group
  const int tm   = blockIdx.y;                             // 32-row group
  if (tn * 64 >= Nc) return;                          // tile-uniform: EXEC all-1s
  const int half = lane >> 4;                         // 0 | 1
  const int l15  = lane & 15;
  const long arow0 = (long)tm * 32 + l15;             // A-frag row (lane%16)
  const long brow0 = (long)tn * 64 + l15;             // B-frag weight row

  // per-lane base pointers (k-invariant)
  const __bf16* pa0 = A + arow0 * lda + (half << 3);
  const __bf16* pa1 = pa0 + 16 * lda;
  const __bf16* pb0 = B + brow0 * ldb + (half << 3);
  const __bf16* pb1 = pb0 + 16 * ldb;
  const __bf16* pb2 = pb0 + 32 * ldb;
  const __bf16* pb3 = pb0 + 48 * ldb;

  v8f acc[2][4];
#pragma unroll
  for (int i = 0; i < 2; ++i)
#pragma unroll
    for (int j = 0; j < 4; ++j) acc[i][j] = (v8f){};

#pragma unroll 2
  for (int k = 0; k < K; k += 32) {
    v16bf af0 = ld_a_frag(pa0 + k);
    v16bf af1 = ld_a_frag(pa1 + k);
    v16bf bf0 = ld_b_frag(pb0 + k);
    v16bf bf1 = ld_b_frag(pb1 + k);
    v16bf bf2 = ld_b_frag(pb2 + k);
    v16bf bf3 = ld_b_frag(pb3 + k);
    acc[0][0] = __builtin_amdgcn_wmma_f32_16x16x32_bf16(false, af0, false, bf0, (short)0, acc[0][0], false, false);
    acc[0][1] = __builtin_amdgcn_wmma_f32_16x16x32_bf16(false, af0, false, bf1, (short)0, acc[0][1], false, false);
    acc[0][2] = __builtin_amdgcn_wmma_f32_16x16x32_bf16(false, af0, false, bf2, (short)0, acc[0][2], false, false);
    acc[0][3] = __builtin_amdgcn_wmma_f32_16x16x32_bf16(false, af0, false, bf3, (short)0, acc[0][3], false, false);
    acc[1][0] = __builtin_amdgcn_wmma_f32_16x16x32_bf16(false, af1, false, bf0, (short)0, acc[1][0], false, false);
    acc[1][1] = __builtin_amdgcn_wmma_f32_16x16x32_bf16(false, af1, false, bf1, (short)0, acc[1][1], false, false);
    acc[1][2] = __builtin_amdgcn_wmma_f32_16x16x32_bf16(false, af1, false, bf2, (short)0, acc[1][2], false, false);
    acc[1][3] = __builtin_amdgcn_wmma_f32_16x16x32_bf16(false, af1, false, bf3, (short)0, acc[1][3], false, false);
  }

  float bv[4];
#pragma unroll
  for (int j = 0; j < 4; ++j)
    bv[j] = (flags & GF_BIAS) ? bias[tn * 64 + j * 16 + l15] : 0.0f;

#pragma unroll
  for (int i = 0; i < 2; ++i) {
    const int m0 = tm * 32 + i * 16 + (half << 3);
#pragma unroll
    for (int j = 0; j < 4; ++j) {
      const int n = tn * 64 + j * 16 + l15;
#pragma unroll
      for (int r = 0; r < 8; ++r) {
        float v = acc[i][j][r] + bv[j];
        if (flags & GF_RELU) v = fmaxf(v, 0.0f);
        long idx = (long)(m0 + r) * ldc + n;
        if (flags & GF_ACC) v += Cf[idx];
        if (flags & GF_OUTBF) Cb[idx] = f2bf(v);
        else                  Cf[idx] = v;
      }
    }
  }
}

// --------------------- LayerNorm + candidate select -----------------------

extern "C" __global__ void k_ln_select(const float* __restrict__ m,
                                       const int* __restrict__ in_deg,
                                       const int* __restrict__ out_deg,
                                       const float* __restrict__ g,
                                       const float* __restrict__ b,
                                       float* __restrict__ xw, int Nn) {
  int n = blockIdx.x * blockDim.y + threadIdx.y;
  if (n >= Nn) return;
  if (!(in_deg[n] == 1 && out_deg[n] == 1)) return;   // non-cand rows keep x
  int lane = threadIdx.x;
  float vals[8], s = 0.f, s2 = 0.f;
#pragma unroll
  for (int j = 0; j < 8; ++j) {
    float v = m[(long)n * 256 + lane + j * 32];
    vals[j] = v; s += v; s2 += v * v;
  }
#pragma unroll
  for (int o = 16; o > 0; o >>= 1) { s += __shfl_xor(s, o); s2 += __shfl_xor(s2, o); }
  float mu  = s * (1.0f / 256.0f);
  float var = s2 * (1.0f / 256.0f) - mu * mu;
  float inv = rsqrtf(var + 1e-5f);
#pragma unroll
  for (int j = 0; j < 8; ++j) {
    int d = lane + j * 32;
    xw[(long)n * 256 + d] = (vals[j] - mu) * inv * g[d] + b[d];
  }
}

// --------------------------- GRU pieces -----------------------------------

// seqs[s,0]=x[call_src[s]]; seqs[s,1+k]=x[call_tgt[s,k]]  (bf16)
extern "C" __global__ void k_build_seq(const float* __restrict__ xw,
                                       const int* __restrict__ csrc,
                                       const int* __restrict__ ctgt,
                                       __bf16* __restrict__ seq, int S, int T) {
  long i = (long)blockIdx.x * blockDim.x + threadIdx.x;
  long tot = (long)S * T * 256;
  if (i >= tot) return;
  int d = (int)(i & 255);
  long st = i >> 8;
  int r = (int)(st % T);
  int s = (int)(st / T);
  int node = (r == 0) ? csrc[s] : ctgt[s * (T - 1) + (r - 1)];
  seq[i] = f2bf(xw[(long)node * 256 + d]);
}

static __device__ __forceinline__ float sigmf(float x) { return 1.0f / (1.0f + expf(-x)); }

// one GRU step: consumes xg[s,t,0:768] (input gates) and gh[s,0:768] (h@Whh^T+bhh)
extern "C" __global__ void k_gru_gate(const float* __restrict__ xg,
                                      const float* __restrict__ gh,
                                      float* __restrict__ h,
                                      __bf16* __restrict__ hbf,
                                      __bf16* __restrict__ ybf, int yld, int yoff,
                                      int t, int T, int S) {
  long i = (long)blockIdx.x * blockDim.x + threadIdx.x;
  if (i >= (long)S * HGDIM) return;
  int s = (int)(i >> 8), j = (int)(i & 255);
  const float* xp = xg + (long)(s * T + t) * G3;
  const float* gp = gh + (long)s * G3;
  float r  = sigmf(xp[j] + gp[j]);
  float z  = sigmf(xp[HGDIM + j] + gp[HGDIM + j]);
  float nn = tanhf(xp[2 * HGDIM + j] + r * gp[2 * HGDIM + j]);
  float hn = (1.0f - z) * nn + z * h[i];
  h[i]   = hn;
  hbf[i] = f2bf(hn);
  if (ybf) ybf[(long)(s * T + t) * yld + yoff + j] = f2bf(hn);
}

// x[call_src[s]] += h2_final[s]   (call_src is unique -> no atomics needed)
extern "C" __global__ void k_add_call(const float* __restrict__ h,
                                      const int* __restrict__ csrc,
                                      float* __restrict__ xw, int S) {
  long i = (long)blockIdx.x * blockDim.x + threadIdx.x;
  if (i >= (long)S * 256) return;
  int s = (int)(i >> 8), d = (int)(i & 255);
  xw[(long)csrc[s] * 256 + d] += h[i];
}

// ----------------------- top-K machinery ----------------------------------

// key = (~bits(||x||^2) << 32) | idx : ascending sort == descending norm,
// lower-index-first on ties (matches jax.lax.top_k).
extern "C" __global__ void k_make_keys(const float* __restrict__ xw,
                                       unsigned long long* __restrict__ keys, int Nn) {
  int n = blockIdx.x * blockDim.y + threadIdx.y;
  if (n >= Nn) return;
  int lane = threadIdx.x;
  float s = 0.f;
#pragma unroll
  for (int j = 0; j < 8; ++j) {
    float v = xw[(long)n * 256 + lane + j * 32];
    s += v * v;
  }
#pragma unroll
  for (int o = 16; o > 0; o >>= 1) s += __shfl_xor(s, o);
  if (lane == 0) {
    unsigned fb = __builtin_bit_cast(unsigned, s);    // >= 0, bits are order-preserving
    keys[n] = ((unsigned long long)(~fb) << 32) | (unsigned)n;
  }
}

extern "C" __global__ void k_bitonic(unsigned long long* __restrict__ keys, int j, int k) {
  int i = blockIdx.x * blockDim.x + threadIdx.x;
  int ixj = i ^ j;
  if (ixj > i) {
    unsigned long long a = keys[i], b = keys[ixj];
    bool asc = ((i & k) == 0);
    if ((a > b) == asc) { keys[i] = b; keys[ixj] = a; }
  }
}

extern "C" __global__ void k_scatter_new(const unsigned long long* __restrict__ keys,
                                         int* __restrict__ noo, int keep) {
  int i = blockIdx.x * blockDim.x + threadIdx.x;
  if (i >= keep) return;
  noo[(int)(keys[i] & 0xFFFFFFFFu)] = i;
}

extern "C" __global__ void k_gather_top(const unsigned long long* __restrict__ keys,
                                        const float* __restrict__ xw,
                                        float* __restrict__ out, int keep) {
  long i = (long)blockIdx.x * blockDim.x + threadIdx.x;
  if (i >= (long)keep * 256) return;
  int r = (int)(i >> 8), d = (int)(i & 255);
  int old = (int)(keys[r] & 0xFFFFFFFFu);
  out[i] = xw[(long)old * 256 + d];
}

extern "C" __global__ void k_remap_edges(const int* __restrict__ row,
                                         const int* __restrict__ col,
                                         const int* __restrict__ etype,
                                         const int* __restrict__ noo,
                                         int* __restrict__ orow, int* __restrict__ ocol,
                                         int* __restrict__ otype, int E) {
  int e = blockIdx.x * blockDim.x + threadIdx.x;
  if (e >= E) return;
  int nr = noo[row[e]], nc = noo[col[e]];
  bool ok = (nr >= 0) && (nc >= 0);
  orow[e]  = ok ? nr : -1;
  ocol[e]  = ok ? nc : -1;
  otype[e] = ok ? etype[e] : -1;
}

// ---------------------------------------------------------------------------

static inline int cdiv(long a, int b) { return (int)((a + b - 1) / b); }

extern "C" void kernel_launch(void* const* d_in, const int* in_sizes, int n_in,
                              void* d_out, int out_size, void* d_ws, size_t ws_size,
                              hipStream_t stream) {
  const float* x     = (const float*)d_in[0];
  const int*   ei    = (const int*)d_in[1];
  const int*   etype = (const int*)d_in[2];
  const int*   csrc  = (const int*)d_in[3];
  const int*   ctgt  = (const int*)d_in[4];
  const float* Wm1   = (const float*)d_in[5];
  const float* bm1   = (const float*)d_in[6];
  const float* Wm2   = (const float*)d_in[7];
  const float* bm2   = (const float*)d_in[8];
  const float* lng   = (const float*)d_in[9];
  const float* lnb   = (const float*)d_in[10];
  const float* aw1   = (const float*)d_in[11];
  const float* ab1   = (const float*)d_in[12];
  const float* aw2   = (const float*)d_in[13];
  const float* ab2   = (const float*)d_in[14];
  const float* g0wih = (const float*)d_in[15];
  const float* g0whh = (const float*)d_in[16];
  const float* g0bih = (const float*)d_in[17];
  const float* g0bhh = (const float*)d_in[18];
  const float* g1wih = (const float*)d_in[19];
  const float* g1whh = (const float*)d_in[20];
  const float* g1bih = (const float*)d_in[21];
  const float* g1bhh = (const float*)d_in[22];
  (void)n_in; (void)ws_size;

  const int Nn = in_sizes[0] / 256;         // nodes
  const int E  = in_sizes[2];               // edges
  const int S  = in_sizes[3];               // call sources
  const int Kc = in_sizes[4] / S;           // targets per source
  const int T  = Kc + 1;                    // GRU sequence length
  const int KEEP = (out_size - 3 * E) / 256;
  int P = 1; while (P < Nn) P <<= 1;        // pow2 pad for bitonic sort

  // ----- workspace layout (256B aligned) -----
  char* base = (char*)d_ws;
  size_t off = 0;
  auto alloc = [&](size_t bytes) { size_t o = off; off = (off + bytes + 255) & ~(size_t)255; return o; };
  size_t o_xw   = alloc((size_t)Nn * 256 * 4);  // working x (f32)
  size_t o_abuf = alloc((size_t)Nn * 512 * 2);  // bf16 GEMM A input
  size_t o_hbuf = alloc((size_t)Nn * 512 * 2);  // bf16 hidden
  size_t o_mbuf = alloc((size_t)Nn * 256 * 4);  // merge MLP output (f32)
  // bf16 weights
  size_t o_wm1  = alloc(512 * 512 * 2);
  size_t o_wm2  = alloc(256 * 512 * 2);
  size_t o_aw1  = alloc(3 * 512 * 256 * 2);
  size_t o_aw2  = alloc(3 * 256 * 512 * 2);
  size_t o_g0wi = alloc(2 * G3 * 256 * 2);
  size_t o_g0wh = alloc(2 * G3 * 256 * 2);
  size_t o_g1wi = alloc(2 * G3 * 512 * 2);
  size_t o_g1wh = alloc(2 * G3 * 256 * 2);
  size_t o_ind  = alloc((size_t)Nn * 4);
  size_t o_outd = alloc((size_t)Nn * 4);
  size_t o_succ = alloc((size_t)Nn * 4);
  size_t o_keys = alloc((size_t)P * 8);
  size_t o_noo  = alloc((size_t)Nn * 4);

  // GRU scratch aliases the (dead-by-then) merge buffers: abuf..mbuf ~302MB
  size_t g = o_abuf;
  auto galloc = [&](size_t bytes) { size_t o = g; g = (g + bytes + 255) & ~(size_t)255; return o; };
  size_t o_seqb = galloc((size_t)S * T * 256 * 2);
  size_t o_xg0  = galloc((size_t)S * T * G3 * 4);   // layer1 fwd input gates
  size_t o_xg1  = galloc((size_t)S * T * G3 * 4);   // layer1 bwd input gates
  size_t o_xg2  = galloc((size_t)S * T * G3 * 4);   // layer2 fwd input gates
  size_t o_gh   = galloc((size_t)S * G3 * 4);
  size_t o_hA   = galloc((size_t)S * 256 * 4);
  size_t o_hB   = galloc((size_t)S * 256 * 4);
  size_t o_hbA  = galloc((size_t)S * 256 * 2);
  size_t o_hbB  = galloc((size_t)S * 256 * 2);
  size_t o_y1b  = galloc((size_t)S * T * 512 * 2);

  float*  xw   = (float*)(base + o_xw);
  __bf16* abuf = (__bf16*)(base + o_abuf);
  __bf16* hbuf = (__bf16*)(base + o_hbuf);
  float*  mbuf = (float*)(base + o_mbuf);
  int* in_deg  = (int*)(base + o_ind);
  int* out_deg = (int*)(base + o_outd);
  int* succ    = (int*)(base + o_succ);
  unsigned long long* keys = (unsigned long long*)(base + o_keys);
  int* noo = (int*)(base + o_noo);

  auto bfw = [&](size_t o) { return (__bf16*)(base + o); };

  const int TB = 256;
  // GEMM launcher: wave computes 32x64; blockDim=(32,2) covers 128 columns.
  auto gemm = [&](const __bf16* A, const __bf16* B, const float* bias,
                  float* Cf, __bf16* Cb, int M, int Nc, int K,
                  int lda, int ldb, int ldc, int flags) {
    dim3 blk(32, 2);
    dim3 grd((unsigned)(Nc / 128), (unsigned)(M / 32));
    k_gemm_wmma<<<grd, blk, 0, stream>>>(A, B, bias, Cf, Cb, M, Nc, K, lda, ldb, ldc, flags);
  };
  auto cvt = [&](const float* in, __bf16* out, long n) {
    k_f32_to_bf16<<<cdiv(n, TB), TB, 0, stream>>>(in, out, n);
  };

  // ---- 0: weights -> bf16 ----
  cvt(Wm1,   bfw(o_wm1),  512 * 512);
  cvt(Wm2,   bfw(o_wm2),  256 * 512);
  cvt(aw1,   bfw(o_aw1),  3 * 512 * 256);
  cvt(aw2,   bfw(o_aw2),  3 * 256 * 512);
  cvt(g0wih, bfw(o_g0wi), 2 * G3 * 256);
  cvt(g0whh, bfw(o_g0wh), 2 * G3 * 256);
  cvt(g1wih, bfw(o_g1wi), 2 * G3 * 512);
  cvt(g1whh, bfw(o_g1wh), 2 * G3 * 256);

  // ---- 1: degrees / successor / working copy ----
  hipMemsetAsync(in_deg, 0, (size_t)Nn * 4, stream);
  hipMemsetAsync(out_deg, 0, (size_t)Nn * 4, stream);
  hipMemsetAsync(succ, 0, (size_t)Nn * 4, stream);   // matches zeros.at[row].set(col)
  k_copy_f32<<<cdiv((long)Nn * 256, TB), TB, 0, stream>>>(x, xw, (long)Nn * 256);
  k_degrees<<<cdiv(E, TB), TB, 0, stream>>>(ei, ei + E, in_deg, out_deg, succ, E);

  // ---- 2: merge MLP (WMMA) + LayerNorm + select ----
  k_build_merge_input<<<cdiv((long)Nn * 256, TB), TB, 0, stream>>>(x, succ, abuf, (long)Nn * 256);
  gemm(abuf, bfw(o_wm1), bm1, nullptr, hbuf, Nn, 512, 512, 512, 512, 512,
       GF_BIAS | GF_RELU | GF_OUTBF);
  gemm(hbuf, bfw(o_wm2), bm2, mbuf, nullptr, Nn, 256, 512, 512, 512, 256, GF_BIAS);
  { dim3 blk(32, 8); k_ln_select<<<cdiv(Nn, 8), blk, 0, stream>>>(mbuf, in_deg, out_deg, lng, lnb, xw, Nn); }

  // ---- 3: residual MLP stack (3 layers, WMMA) ----
  for (int i = 0; i < 3; ++i) {
    cvt(xw, abuf, (long)Nn * 256);  // abuf reused as [N,256] bf16
    gemm(abuf, bfw(o_aw1) + (size_t)i * 512 * 256, ab1 + i * 512,
         nullptr, hbuf, Nn, 512, 256, 256, 256, 512, GF_BIAS | GF_RELU | GF_OUTBF);
    gemm(hbuf, bfw(o_aw2) + (size_t)i * 256 * 512, ab2 + i * 256,
         xw, nullptr, Nn, 256, 512, 512, 512, 256, GF_BIAS | GF_ACC);
  }

  // ---- 4: BiGRU over call sequences ----
  __bf16* seqb = (__bf16*)(base + o_seqb);
  float*  xgd[2] = { (float*)(base + o_xg0), (float*)(base + o_xg1) };
  float*  xg2  = (float*)(base + o_xg2);
  float*  ghb  = (float*)(base + o_gh);
  float*  hA   = (float*)(base + o_hA);
  float*  hB   = (float*)(base + o_hB);
  __bf16* hbA  = (__bf16*)(base + o_hbA);
  __bf16* hbB  = (__bf16*)(base + o_hbB);
  __bf16* y1b  = (__bf16*)(base + o_y1b);

  k_build_seq<<<cdiv((long)S * T * 256, TB), TB, 0, stream>>>(xw, csrc, ctgt, seqb, S, T);

  // layer-1 input gate GEMMs (both directions)
  for (int d = 0; d < 2; ++d)
    gemm(seqb, bfw(o_g0wi) + (size_t)d * G3 * 256, g0bih + d * G3,
         xgd[d], nullptr, S * T, G3, 256, 256, 256, G3, GF_BIAS);

  hipMemsetAsync(hA, 0, (size_t)S * 256 * 4, stream);
  hipMemsetAsync(hB, 0, (size_t)S * 256 * 4, stream);
  hipMemsetAsync(hbA, 0, (size_t)S * 256 * 2, stream);
  hipMemsetAsync(hbB, 0, (size_t)S * 256 * 2, stream);

  const int gateGrid = cdiv((long)S * 256, TB);
  for (int t = 0; t < T; ++t) {
    // forward direction, step t
    gemm(hbA, bfw(o_g0wh) + 0, g0bhh + 0, ghb, nullptr, S, G3, 256, 256, 256, G3, GF_BIAS);
    k_gru_gate<<<gateGrid, TB, 0, stream>>>(xgd[0], ghb, hA, hbA, y1b, 512, 0, t, T, S);
    // backward direction, step T-1-t (outputs aligned to input order)
    gemm(hbB, bfw(o_g0wh) + (size_t)G3 * 256, g0bhh + G3, ghb, nullptr, S, G3, 256, 256, 256, G3, GF_BIAS);
    k_gru_gate<<<gateGrid, TB, 0, stream>>>(xgd[1], ghb, hB, hbB, y1b, 512, 256, T - 1 - t, T, S);
  }

  // layer-2: only the FORWARD direction is consumed (h2[:, -1, :D], D==HG)
  gemm(y1b, bfw(o_g1wi) + 0, g1bih + 0, xg2, nullptr, S * T, G3, 512, 512, 512, G3, GF_BIAS);
  hipMemsetAsync(hA, 0, (size_t)S * 256 * 4, stream);
  hipMemsetAsync(hbA, 0, (size_t)S * 256 * 2, stream);
  for (int t = 0; t < T; ++t) {
    gemm(hbA, bfw(o_g1wh) + 0, g1bhh + 0, ghb, nullptr, S, G3, 256, 256, 256, G3, GF_BIAS);
    k_gru_gate<<<gateGrid, TB, 0, stream>>>(xg2, ghb, hA, hbA, nullptr, 0, 0, t, T, S);
  }
  k_add_call<<<gateGrid, TB, 0, stream>>>(hA, csrc, xw, S);

  // ---- 5: top-K compression ----
  hipMemsetAsync(keys, 0xFF, (size_t)P * 8, stream);  // padding sorts last
  { dim3 blk(32, 8); k_make_keys<<<cdiv(Nn, 8), blk, 0, stream>>>(xw, keys, Nn); }
  for (int kk = 2; kk <= P; kk <<= 1)
    for (int j = kk >> 1; j > 0; j >>= 1)
      k_bitonic<<<P / TB, TB, 0, stream>>>(keys, j, kk);

  hipMemsetAsync(noo, 0xFF, (size_t)Nn * 4, stream);  // -1
  k_scatter_new<<<cdiv(KEEP, TB), TB, 0, stream>>>(keys, noo, KEEP);

  float* outf = (float*)d_out;                         // x[top_idx]
  int*   oint = (int*)(outf + (size_t)KEEP * 256);     // int32 outputs stored as raw bits
  k_gather_top<<<cdiv((long)KEEP * 256, TB), TB, 0, stream>>>(keys, xw, outf, KEEP);
  k_remap_edges<<<cdiv(E, TB), TB, 0, stream>>>(ei, ei + E, etype, noo,
                                                oint, oint + E, oint + 2 * E, E);
}
